// GRAFF_62783831933371
// MI455X (gfx1250) — compile-verified, matrix operand order checked by reading, not run on hardware
//
#include <hip/hip_runtime.h>
#include <hip/hip_bf16.h>

typedef __bf16 bh16;
typedef __attribute__((ext_vector_type(16))) __bf16 v16bf;
typedef __attribute__((ext_vector_type(8)))  float  v8f;

#define GH       256     // HIDDEN
#define BM       128     // block M tile (8 waves x 16 rows)
#define KSTEP    32
#define LDSB     40      // LDS row stride in bf16 elems (32 data + 8 pad, 80B: 16B aligned)

#if __has_builtin(__builtin_amdgcn_sched_barrier)
#define SCHED_FENCE() __builtin_amdgcn_sched_barrier(0)
#else
#define SCHED_FENCE()
#endif

__device__ __forceinline__ float elu1(float x) {
  return x > 0.0f ? x : (__expf(x) - 1.0f);
}

// ---------------------------------------------------------------------------
// Unified WMMA GEMM:  C[M x 256] = A[M x K](fp32) * B[256 x Kp](bf16)^T
// optional bias + ELU epilogue.  256 threads = 8 waves; wave tile 16 x 256.
// Double-buffered LDS weight tile.  B fragments software-pipelined two-deep
// (bE/bO rotation + sched fences) so each WMMA runs with the next tile's
// ds_loads in flight (s_wait_dscnt 2 instead of 0).
// ---------------------------------------------------------------------------
__global__ __launch_bounds__(256)
void graff_gemm256(const float* __restrict__ A, const bh16* __restrict__ B,
                   float* __restrict__ C, int M, int K, int Kp,
                   const float* __restrict__ bias, int act) {
  __shared__ bh16 sB[2][GH * LDSB];

  const int tid  = threadIdx.x;
  const int lane = tid & 31;
  const int wave = tid >> 5;
  const int r    = lane & 15;          // N index inside tile (B/C), row inside A tile
  const int hi   = (lane >> 4) & 1;    // K-chunk selector per ISA 16-bit layouts
  const int mBase = blockIdx.x * BM + wave * 16;

  v8f acc[16];
  const v8f vzero = {0.f, 0.f, 0.f, 0.f, 0.f, 0.f, 0.f, 0.f};
#pragma unroll
  for (int t = 0; t < 16; ++t) acc[t] = vzero;

  const int  mA  = mBase + r;
  const bool inM = (mA < M);
  const float* Arow = A + (size_t)(inM ? mA : 0) * (size_t)K;

  const int nIter = Kp / KSTEP;
  const bh16* Brow = B + (size_t)tid * (size_t)Kp;   // thread tid stages row tid

  // prologue: global fetch + LDS store of tile 0 into buffer 0
  uint4 st[4];
#pragma unroll
  for (int i = 0; i < 4; ++i) st[i] = ((const uint4*)Brow)[i];
  {
    uint4* d = (uint4*)(&sB[0][tid * LDSB]);
#pragma unroll
    for (int i = 0; i < 4; ++i) d[i] = st[i];
  }
  __syncthreads();

  for (int it = 0; it < nIter; ++it) {
    const int k0  = it * KSTEP;
    const int cur = it & 1;
    const bool more = (it + 1 < nIter);

    // issue global fetch of the NEXT B tile early (hidden under the WMMAs)
    if (more) {
      const uint4* s = (const uint4*)(Brow + (k0 + KSTEP));
#pragma unroll
      for (int i = 0; i < 4; ++i) st[i] = s[i];
      __builtin_prefetch(Arow + k0 + KSTEP, 0, 3);   // global_prefetch next A chunk
    }

    // A fragment: fp32 -> bf16 in registers, ISA 16-bit A layout
    v16bf a;
    const int kA = k0 + hi * 8;
    if (k0 + KSTEP <= K) {
#pragma unroll
      for (int i = 0; i < 8; ++i) {
        a[i]     = (bh16)Arow[kA + i];
        a[i + 8] = (bh16)Arow[kA + 16 + i];
      }
    } else {  // K tail (encoder: K=2613)
#pragma unroll
      for (int i = 0; i < 8; ++i) {
        const int k1 = kA + i, k2 = kA + 16 + i;
        a[i]     = (bh16)((k1 < K) ? Arow[k1] : 0.0f);
        a[i + 8] = (bh16)((k2 < K) ? Arow[k2] : 0.0f);
      }
    }

    // 16 WMMAs, two-deep software pipeline on B fragments.
    const bh16* base = &sB[cur][r * LDSB + hi * 8];
#define LOADB(frag, t_)                                                    \
    {                                                                      \
      const bh16* bp_ = base + (t_) * (16 * LDSB);                         \
      ((uint4*)&(frag))[0] = *(const uint4*)(bp_);                         \
      ((uint4*)&(frag))[1] = *(const uint4*)(bp_ + 16);                    \
    }
    v16bf bE, bO;
    LOADB(bE, 0);
    LOADB(bO, 1);
    SCHED_FENCE();
#pragma unroll
    for (int t = 0; t < 16; t += 2) {
      acc[t] = __builtin_amdgcn_wmma_f32_16x16x32_bf16(
          false, a, false, bE, (short)0, acc[t], false, false);
      if (t + 2 < 16) LOADB(bE, t + 2);
      SCHED_FENCE();
      acc[t + 1] = __builtin_amdgcn_wmma_f32_16x16x32_bf16(
          false, a, false, bO, (short)0, acc[t + 1], false, false);
      if (t + 3 < 16) LOADB(bO, t + 3);
      SCHED_FENCE();
    }
#undef LOADB

    // stage the next tile into the other LDS buffer, then one barrier per step
    if (more) {
      uint4* d = (uint4*)(&sB[1 - cur][tid * LDSB]);
#pragma unroll
      for (int i = 0; i < 4; ++i) d[i] = st[i];
    }
    __syncthreads();
  }

  // epilogue: C/D layout -> VGPR j is M=j (lanes 0-15) / M=j+8 (lanes 16-31)
#pragma unroll
  for (int t = 0; t < 16; ++t) {
    const int n  = t * 16 + r;
    const float bv = bias ? bias[n] : 0.0f;
#pragma unroll
    for (int j = 0; j < 8; ++j) {
      const int m = mBase + j + hi * 8;
      if (m < M) {
        float v = acc[t][j] + bv;
        if (act) v = elu1(v);
        C[(size_t)m * GH + n] = v;
      }
    }
  }
}

// ---------------------------------------------------------------------------
// Small helper kernels
// ---------------------------------------------------------------------------
__global__ void graff_zero(float* __restrict__ p, long n) {
  const long stride = (long)gridDim.x * blockDim.x;
  for (long i = (long)blockIdx.x * blockDim.x + threadIdx.x; i < n; i += stride)
    p[i] = 0.0f;
}

__global__ void graff_cvt_pad(const float* __restrict__ src, bh16* __restrict__ dst,
                              int N, int K, int Kp) {
  const long n = (long)N * Kp;
  const long stride = (long)gridDim.x * blockDim.x;
  for (long i = (long)blockIdx.x * blockDim.x + threadIdx.x; i < n; i += stride) {
    const int rr = (int)(i / Kp), k = (int)(i % Kp);
    dst[i] = (bh16)((k < K) ? src[(size_t)rr * K + k] : 0.0f);
  }
}

// Wp = triu(W[:,:-2],1) symmetrized + diag(q*rowabssum + r); store bf16
__global__ void graff_build_wp(const float* __restrict__ W, bh16* __restrict__ Wp) {
  const int i = threadIdx.x;            // 256 threads, one row each
  const int LD = GH + 2;
  float asum = 0.0f;
  for (int j = 0; j < GH; ++j) {
    float v;
    if (i < j)      v = W[(size_t)i * LD + j];
    else if (i > j) v = W[(size_t)j * LD + i];
    else            v = 0.0f;
    asum += fabsf(v);
    Wp[(size_t)i * GH + j] = (bh16)v;
  }
  const float q = W[(size_t)i * LD + GH];
  const float rr = W[(size_t)i * LD + GH + 1];
  Wp[(size_t)i * GH + i] = (bh16)(q * asum + rr);
}

__global__ __launch_bounds__(256)
void graff_bn_stats(const float* __restrict__ h, float* __restrict__ sums, int M) {
  const int c = threadIdx.x;            // channel
  float s = 0.0f, s2 = 0.0f;
  for (int m = blockIdx.x; m < M; m += gridDim.x) {
    const float v = h[(size_t)m * GH + c];
    s += v; s2 += v * v;
  }
  atomicAdd(&sums[c], s);
  atomicAdd(&sums[GH + c], s2);
}

__global__ void graff_bn_apply(float* __restrict__ h, float* __restrict__ h0,
                               const float* __restrict__ sums,
                               const float* __restrict__ gamma,
                               const float* __restrict__ beta, int M) {
  const long n = (long)M * GH;
  const long stride = (long)gridDim.x * blockDim.x;
  const float invN = 1.0f / (float)M;
  for (long i = (long)blockIdx.x * blockDim.x + threadIdx.x; i < n; i += stride) {
    const int c = (int)(i & (GH - 1));
    const float mean = sums[c] * invN;
    const float var  = sums[GH + c] * invN - mean * mean;
    const float sc   = gamma[c] * rsqrtf(var + 1e-5f);
    const float v = (h[i] - mean) * sc + beta[c];
    h[i] = v;
    h0[i] = v;
  }
}

__global__ void graff_deg_count(const int* __restrict__ col, float* __restrict__ deg, int nE) {
  const int stride = gridDim.x * blockDim.x;
  for (int e = blockIdx.x * blockDim.x + threadIdx.x; e < nE; e += stride)
    atomicAdd(&deg[col[e]], 1.0f);
}

__global__ void graff_deg_inv(float* __restrict__ deg, int N) {
  const int stride = gridDim.x * blockDim.x;
  for (int i = blockIdx.x * blockDim.x + threadIdx.x; i < N; i += stride) {
    const float d = deg[i];
    deg[i] = (d > 0.0f) ? rsqrtf(d) : 0.0f;
  }
}

__global__ void graff_edge_norm(const int* __restrict__ row, const int* __restrict__ col,
                                const float* __restrict__ dinv, float* __restrict__ norm,
                                int nE) {
  const int stride = gridDim.x * blockDim.x;
  for (int e = blockIdx.x * blockDim.x + threadIdx.x; e < nE; e += stride)
    norm[e] = dinv[row[e]] * dinv[col[e]];
}

// one block per edge: 256 channels scatter-added with global_atomic_add_f32
__global__ __launch_bounds__(256)
void graff_scatter(const float* __restrict__ outp, const int* __restrict__ row,
                   const int* __restrict__ col, const float* __restrict__ norm,
                   float* __restrict__ agg, int nE) {
  const int c = threadIdx.x;
  for (int e = blockIdx.x; e < nE; e += gridDim.x) {
    const int rs = row[e];
    const int cd = col[e];
    const float w = norm[e];
    atomicAdd(&agg[(size_t)cd * GH + c], w * outp[(size_t)rs * GH + c]);
  }
}

__global__ void graff_update(float* __restrict__ h, const float* __restrict__ h0,
                             const float* __restrict__ agg, const float* __restrict__ ext_w,
                             const float* __restrict__ src_b, long n) {
  const long stride = (long)gridDim.x * blockDim.x;
  const float sb = src_b[0];
  for (long i = (long)blockIdx.x * blockDim.x + threadIdx.x; i < n; i += stride) {
    const int c = (int)(i & (GH - 1));
    const float out = agg[i] - h[i] * ext_w[c] - sb * h0[i];
    h[i] = h[i] + 0.1f * elu1(out);
  }
}

// ---------------------------------------------------------------------------
extern "C" void kernel_launch(void* const* d_in, const int* in_sizes, int n_in,
                              void* d_out, int out_size, void* d_ws, size_t ws_size,
                              hipStream_t stream) {
  const int M = 100000, K_IN = 2613, NE = 320000;
  const int KP_IN = (K_IN + 31) & ~31;   // 2624

  const float* x     = (const float*)d_in[0];
  const int*   ei    = (const int*)d_in[1];
  const int*   row   = ei;
  const int*   col   = ei + NE;
  const float* enc_w = (const float*)d_in[2];
  const float* gamma = (const float*)d_in[3];
  const float* beta  = (const float*)d_in[4];
  const float* ext_w = (const float*)d_in[5];
  const float* src_b = (const float*)d_in[6];
  const float* pw_W  = (const float*)d_in[7];
  const float* l1w   = (const float*)d_in[8];
  const float* l1b   = (const float*)d_in[9];
  const float* l2w   = (const float*)d_in[10];
  const float* l2b   = (const float*)d_in[11];

  char* wsp = (char*)d_ws;
  auto carve = [&](size_t bytes) -> char* {
    char* p = wsp;
    wsp += (bytes + 255) & ~(size_t)255;
    return p;
  };
  float* h    = (float*)carve((size_t)M * GH * 4);
  float* h0   = (float*)carve((size_t)M * GH * 4);
  float* agg  = (float*)carve((size_t)M * GH * 4);
  bh16* encwb = (bh16*)carve((size_t)GH * KP_IN * 2);
  bh16* wpb   = (bh16*)carve((size_t)GH * GH * 2);
  bh16* l1wb  = (bh16*)carve((size_t)GH * GH * 2);
  bh16* l2wb  = (bh16*)carve((size_t)GH * GH * 2);
  float* sums = (float*)carve(2 * GH * 4);
  float* deg  = (float*)carve((size_t)M * 4);
  float* norm = (float*)carve((size_t)NE * 4);
  float* outp = (float*)d_out;   // reused as per-layer out_p scratch

  const int gGemm = (M + BM - 1) / BM;

  // prep
  graff_zero<<<8, 256, 0, stream>>>(sums, 2 * GH);
  graff_zero<<<512, 256, 0, stream>>>(deg, M);
  graff_cvt_pad<<<1024, 256, 0, stream>>>(enc_w, encwb, GH, K_IN, KP_IN);
  graff_cvt_pad<<<256, 256, 0, stream>>>(l1w, l1wb, GH, GH, GH);
  graff_cvt_pad<<<256, 256, 0, stream>>>(l2w, l2wb, GH, GH, GH);
  graff_build_wp<<<1, 256, 0, stream>>>(pw_W, wpb);

  // encoder + batchnorm
  graff_gemm256<<<gGemm, 256, 0, stream>>>(x, encwb, h, M, K_IN, KP_IN, nullptr, 0);
  graff_bn_stats<<<256, 256, 0, stream>>>(h, sums, M);
  graff_bn_apply<<<4096, 256, 0, stream>>>(h, h0, sums, gamma, beta, M);

  // graph normalization coefficients
  graff_deg_count<<<1024, 256, 0, stream>>>(col, deg, NE);
  graff_deg_inv<<<512, 256, 0, stream>>>(deg, M);
  graff_edge_norm<<<1024, 256, 0, stream>>>(row, col, deg, norm, NE);

  // 4 GRAFF layers
  for (int layer = 0; layer < 4; ++layer) {
    graff_gemm256<<<gGemm, 256, 0, stream>>>(h, wpb, outp, M, GH, GH, nullptr, 0);
    graff_zero<<<8192, 256, 0, stream>>>(agg, (long)M * GH);
    graff_scatter<<<NE, 256, 0, stream>>>(outp, row, col, norm, agg, NE);
    graff_update<<<8192, 256, 0, stream>>>(h, h0, agg, ext_w, src_b, (long)M * GH);
  }

  // head: lin1 (bias+ELU) -> agg, lin2 (bias) -> d_out
  graff_gemm256<<<gGemm, 256, 0, stream>>>(h, l1wb, agg, M, GH, GH, l1b, 1);
  graff_gemm256<<<gGemm, 256, 0, stream>>>(agg, l2wb, outp, M, GH, GH, l2b, 0);
}
